// GraphSAGEModel_24197845745727
// MI455X (gfx1250) — compile-verified, matrix operand order checked by reading, not run on hardware
//
#include <hip/hip_runtime.h>
#include <hip/hip_fp16.h>

// ---------------------------------------------------------------------------
// GraphSAGE forward for MI455X (gfx1250, wave32, WMMA, pk_add_f16 atomics).
//   layer: h = relu([mean_aggr(x) | x] @ [wl;wr]^T_packed + b)   (K=256 GEMM)
//   pool : g = segment_mean(h2, batch);  out = log_softmax(g@wf + bf)
// Node features flow as f16 end-to-end; accumulation in f32 where it matters
// (WMMA C/D, degree counts, graph pooling). GEMM inner loop double-buffers
// B fragments so LDS loads pipeline ahead of the matrix unit.
// ---------------------------------------------------------------------------

#define GN 50000      // nodes
#define GE 800000     // edges
#define GF 128        // feature dim (both layers)
#define GH 128        // hidden
#define GC 16         // classes
#define GG 512        // graphs

typedef __attribute__((ext_vector_type(16))) _Float16 v16h;
typedef __attribute__((ext_vector_type(8)))  _Float16 v8h;
typedef __attribute__((ext_vector_type(8)))  float    v8f;

// constant modifier args via template => always ICE for the builtin
template <bool RA>
static __device__ __forceinline__ v8f wmma_f16(v16h a, v16h b, v8f c) {
    return __builtin_amdgcn_wmma_f32_16x16x32_f16(
        false, a, false, b, (short)0, c, RA, false);
}

// ---------------------------------------------------------------- zero fill
__global__ void sage_zero_f32(float* __restrict__ p, long long n) {
    long long i = (long long)blockIdx.x * blockDim.x + threadIdx.x;
    long long stride = (long long)gridDim.x * blockDim.x;
    for (; i < n; i += stride) p[i] = 0.0f;
}

// ------------------------------------------------------ f32 -> f16 convert
__global__ void sage_cvt_f16(const float* __restrict__ in,
                             _Float16* __restrict__ out, long long n) {
    long long i = (long long)blockIdx.x * blockDim.x + threadIdx.x;
    long long stride = (long long)gridDim.x * blockDim.x;
    for (; i < n; i += stride) out[i] = (_Float16)in[i];
}

// --------------------------------------------- pack weights: Wt[n][k] (f16)
// Wt[n*256 + k] = k<128 ? wl[k][n] : wr[k-128][n]  -> a B-fragment lane reads
// 8 contiguous K values = one 16B chunk.
__global__ void sage_pack_w(const float* __restrict__ wl,
                            const float* __restrict__ wr,
                            _Float16* __restrict__ wt) {
    int i = blockIdx.x * blockDim.x + threadIdx.x;      // 0 .. 128*256-1
    int total = GH * 256;
    for (; i < total; i += gridDim.x * blockDim.x) {
        int n = i >> 8;         // output column
        int k = i & 255;        // concat-K
        float v = (k < GF) ? wl[k * GH + n] : wr[(k - GF) * GH + n];
        wt[i] = (_Float16)v;
    }
}

// ------------------------------------------- edge scatter, packed f16 atomics
// one thread: one edge x 8 features = 4 global_atomic_pk_add_f16 ops.
__global__ void sage_scatter(const _Float16* __restrict__ xh,
                             const int* __restrict__ src,
                             const int* __restrict__ dst,
                             _Float16* __restrict__ msgh,
                             float* __restrict__ cnt) {
    long long t = (long long)blockIdx.x * blockDim.x + threadIdx.x;
    long long total = (long long)GE * 16;               // 16 chunks of 8 halves
    long long stride = (long long)gridDim.x * blockDim.x;
    for (; t < total; t += stride) {
        int e = (int)(t >> 4);
        int p = (int)(t & 15);
        int s = src[e], d = dst[e];
        const __half2* xs = (const __half2*)(xh + (long long)s * GF) + p * 4;
        __half2*       ms = (__half2*)(msgh + (long long)d * GF) + p * 4;
        atomicAdd(ms + 0, xs[0]);
        atomicAdd(ms + 1, xs[1]);
        atomicAdd(ms + 2, xs[2]);
        atomicAdd(ms + 3, xs[3]);
        if (p == 0) atomicAdd(&cnt[d], 1.0f);
    }
}

// ---------------------------------- build A = [msg/cnt | x]  as f16 [N,256]
__global__ void sage_build_a(const _Float16* __restrict__ msgh,
                             const float* __restrict__ cnt,
                             const _Float16* __restrict__ xh,
                             _Float16* __restrict__ A) {
    long long i = (long long)blockIdx.x * blockDim.x + threadIdx.x;
    long long total = (long long)GN * 256;
    long long stride = (long long)gridDim.x * blockDim.x;
    for (; i < total; i += stride) {
        long long r = i >> 8;
        int k = (int)(i & 255);
        _Float16 v;
        if (k < GF) {
            float inv = 1.0f / fmaxf(cnt[r], 1.0f);
            v = (_Float16)((float)msgh[r * GF + k] * inv);
        } else {
            v = xh[r * GF + (k - GF)];
        }
        A[i] = v;
    }
}

// --------------------------------------------------- fused WMMA GEMM + relu
// out[r][n] = relu( sum_{k<256} A[r][k] * Wt[n][k] + bias[n] )  (f16 out)
// block = 256 threads = 8 waves; block tile = 128 rows x 128 cols;
// wave tile = 16 rows x 128 cols (8 f32 accumulators, double-buffered B).
__global__ void __launch_bounds__(256)
sage_gemm_wmma(const _Float16* __restrict__ A,
               const _Float16* __restrict__ Wt,
               const float* __restrict__ bias,
               _Float16* __restrict__ out) {
    __shared__ _Float16 sW[GH * 256];                   // 64 KB of WGP LDS

    // cooperative weight stage: 4096 x v8h chunks, 16 per thread
    {
        const v8h* srcv = (const v8h*)Wt;
        v8h* dstv = (v8h*)sW;
        int t = threadIdx.x;
        #pragma unroll
        for (int i = 0; i < 16; ++i) dstv[t + i * 256] = srcv[t + i * 256];
    }
    __syncthreads();

    const int wave = threadIdx.x >> 5;
    const int lane = threadIdx.x & 31;
    const int hs   = lane >> 4;          // 0: K+0..7/K+16..23, 1: +8 shifted
    const int l15  = lane & 15;

    const long long rowBase = (long long)blockIdx.x * 128 + wave * 16;
    long long aRow = rowBase + l15;
    if (aRow > GN - 1) aRow = GN - 1;    // clamp OOB tail reads
    const _Float16* Arow = A + aRow * 256;

    v8f acc[8];
    #pragma unroll
    for (int nt = 0; nt < 8; ++nt) acc[nt] = (v8f)0.0f;

    #pragma unroll
    for (int ks = 0; ks < 8; ++ks) {     // K = 256, 32 per WMMA
        const int kA = ks * 32 + hs * 8;

        v16h afrag;
        {
            v8h* ap = (v8h*)&afrag;
            ap[0] = *(const v8h*)(Arow + kA);
            ap[1] = *(const v8h*)(Arow + kA + 16);
        }

        // B-fragment fetch from LDS (16-bit B layout: lane = col, two 16B
        // K-chunks per lane)
        auto loadB = [&](int nt) -> v16h {
            const _Float16* wp = &sW[(nt * 16 + l15) * 256 + kA];
            v16h b;
            v8h* bp = (v8h*)&b;
            bp[0] = *(const v8h*)(wp);
            bp[1] = *(const v8h*)(wp + 16);
            return b;
        };

        // software-pipelined: load tile nt+1 before issuing wmma on tile nt
        v16h b0 = loadB(0);
        v16h b1 = loadB(1);
        acc[0] = wmma_f16<false>(afrag, b0, acc[0]);
        b0 = loadB(2);
        acc[1] = wmma_f16<false>(afrag, b1, acc[1]);
        b1 = loadB(3);
        acc[2] = wmma_f16<false>(afrag, b0, acc[2]);
        b0 = loadB(4);
        acc[3] = wmma_f16<false>(afrag, b1, acc[3]);
        b1 = loadB(5);
        acc[4] = wmma_f16<false>(afrag, b0, acc[4]);
        b0 = loadB(6);
        acc[5] = wmma_f16<false>(afrag, b1, acc[5]);
        b1 = loadB(7);
        acc[6] = wmma_f16<false>(afrag, b0, acc[6]);
        acc[7] = wmma_f16<false>(afrag, b1, acc[7]);
    }

    // epilogue: D layout -> lane = col, vgpr v = row (m = v + 8*hs)
    #pragma unroll
    for (int nt = 0; nt < 8; ++nt) {
        const int col = nt * 16 + l15;
        const float bv = bias[col];
        #pragma unroll
        for (int v = 0; v < 8; ++v) {
            long long row = rowBase + v + hs * 8;
            if (row < GN) {
                float val = acc[nt][v] + bv;
                out[row * GH + col] = (_Float16)(val > 0.0f ? val : 0.0f);
            }
        }
    }
}

// ------------------------------------------------------- global mean pool
__global__ void sage_pool(const _Float16* __restrict__ h,
                          const int* __restrict__ batch,
                          float* __restrict__ gsum,
                          float* __restrict__ gcnt) {
    long long t = (long long)blockIdx.x * blockDim.x + threadIdx.x;
    long long total = (long long)GN * 32;               // 4 feats per thread
    long long stride = (long long)gridDim.x * blockDim.x;
    for (; t < total; t += stride) {
        int r = (int)(t >> 5);
        int p = (int)(t & 31);
        int b = batch[r];
        const _Float16* hs = h + (long long)r * GH + p * 4;
        float* gs = gsum + (long long)b * GH + p * 4;
        atomicAdd(gs + 0, (float)hs[0]);
        atomicAdd(gs + 1, (float)hs[1]);
        atomicAdd(gs + 2, (float)hs[2]);
        atomicAdd(gs + 3, (float)hs[3]);
        if (p == 0) atomicAdd(&gcnt[b], 1.0f);
    }
}

// --------------------------------------- logits + log_softmax (one wave/graph)
__global__ void sage_final(const float* __restrict__ gsum,
                           const float* __restrict__ gcnt,
                           const float* __restrict__ wf,
                           const float* __restrict__ bf,
                           float* __restrict__ out) {
    const int g = blockIdx.x;            // 512 graphs
    const int lane = threadIdx.x;        // 32 lanes, 16 active for classes
    const float inv = 1.0f / fmaxf(gcnt[g], 1.0f);

    float logit = 0.0f;
    if (lane < GC) {
        logit = bf[lane];
        const float* gr = gsum + (long long)g * GH;
        for (int k = 0; k < GH; ++k)
            logit += (gr[k] * inv) * wf[k * GC + lane];
    }
    // log-softmax across the 16 class lanes
    float m = logit;
    for (int o = 8; o >= 1; o >>= 1) m = fmaxf(m, __shfl_xor(m, o, 16));
    float ex = __expf(logit - m);
    float s = ex;
    for (int o = 8; o >= 1; o >>= 1) s += __shfl_xor(s, o, 16);
    if (lane < GC) out[(long long)g * GC + lane] = logit - m - __logf(s);
}

// ---------------------------------------------------------------------------
extern "C" void kernel_launch(void* const* d_in, const int* in_sizes, int n_in,
                              void* d_out, int out_size, void* d_ws, size_t ws_size,
                              hipStream_t stream) {
    (void)in_sizes; (void)n_in; (void)out_size; (void)ws_size;

    const float* x   = (const float*)d_in[0];
    const int*   ei  = (const int*)d_in[1];
    const int*   bat = (const int*)d_in[2];
    const float* w1l = (const float*)d_in[3];
    const float* b1  = (const float*)d_in[4];
    const float* w1r = (const float*)d_in[5];
    const float* w2l = (const float*)d_in[6];
    const float* b2  = (const float*)d_in[7];
    const float* w2r = (const float*)d_in[8];
    const float* wf  = (const float*)d_in[9];
    const float* bf  = (const float*)d_in[10];
    float* outp = (float*)d_out;

    const int* src = ei;
    const int* dst = ei + GE;

    // workspace carve-up (256B aligned)
    char* base = (char*)d_ws;
    size_t off = 0;
    auto take = [&](size_t bytes) -> void* {
        void* p = base + off;
        off = (off + bytes + 255) & ~(size_t)255;
        return p;
    };
    _Float16*  xh   = (_Float16*) take((size_t)GN * GF * 2);
    _Float16*  msgh = (_Float16*) take((size_t)GN * GF * 2);
    float*     cnt  = (float*)    take((size_t)GN * 4);
    _Float16*  Abuf = (_Float16*) take((size_t)GN * 256 * 2);
    _Float16*  h1h  = (_Float16*) take((size_t)GN * GH * 2);
    _Float16*  h2h  = (_Float16*) take((size_t)GN * GH * 2);
    _Float16*  wt1  = (_Float16*) take((size_t)GH * 256 * 2);
    _Float16*  wt2  = (_Float16*) take((size_t)GH * 256 * 2);
    float*     gsum = (float*)    take((size_t)GG * GH * 4);
    float*     gcnt = (float*)    take((size_t)GG * 4);

    const int ZB = 2048, TB = 256;
    const int gemmBlocks = (GN + 127) / 128;     // 391

    // one-time prep: weight packing + input conversion
    sage_pack_w<<<128, TB, 0, stream>>>(w1l, w1r, wt1);
    sage_pack_w<<<128, TB, 0, stream>>>(w2l, w2r, wt2);
    sage_cvt_f16<<<4096, TB, 0, stream>>>(x, xh, (long long)GN * GF);

    // ---- layer 1
    sage_zero_f32<<<ZB, TB, 0, stream>>>((float*)msgh, (long long)GN * GF / 2);
    sage_zero_f32<<<ZB, TB, 0, stream>>>(cnt, (long long)GN);
    sage_scatter<<<8192, TB, 0, stream>>>(xh, src, dst, msgh, cnt);
    sage_build_a<<<4096, TB, 0, stream>>>(msgh, cnt, xh, Abuf);
    sage_gemm_wmma<<<gemmBlocks, TB, 0, stream>>>(Abuf, wt1, b1, h1h);

    // ---- layer 2
    sage_zero_f32<<<ZB, TB, 0, stream>>>((float*)msgh, (long long)GN * GF / 2);
    sage_zero_f32<<<ZB, TB, 0, stream>>>(cnt, (long long)GN);
    sage_scatter<<<8192, TB, 0, stream>>>(h1h, src, dst, msgh, cnt);
    sage_build_a<<<4096, TB, 0, stream>>>(msgh, cnt, h1h, Abuf);
    sage_gemm_wmma<<<gemmBlocks, TB, 0, stream>>>(Abuf, wt2, b2, h2h);

    // ---- pooling + classifier
    sage_zero_f32<<<ZB, TB, 0, stream>>>(gsum, (long long)GG * GH);
    sage_zero_f32<<<ZB, TB, 0, stream>>>(gcnt, (long long)GG);
    sage_pool<<<2048, TB, 0, stream>>>(h2h, bat, gsum, gcnt);
    sage_final<<<GG, 32, 0, stream>>>(gsum, gcnt, wf, bf, outp);
}
